// GCN_31954556683004
// MI455X (gfx1250) — matrix operand order repaired and checked by reading, not compiled, not run on hardware
//
#include <hip/hip_runtime.h>
#include <hip/hip_bf16.h>

// ---------------------------------------------------------------------------
// GraphConv x3 + linear head for gfx1250 (MI455X).
//   layer: h' = relu( aggr @ w_rel^T + b_rel + h @ w_root^T )
//   fused GEMM:  A = [aggr | h]  (N x 256),  B = [w_rel^T ; w_root^T] (256 x 128)
//   matrix op:   v_wmma_f32_16x16x32_bf16 (bf16 in, f32 accumulate)
// ---------------------------------------------------------------------------

typedef __attribute__((ext_vector_type(16))) __bf16 v16bf;
typedef __attribute__((ext_vector_type(8)))  float  v8f;
typedef __attribute__((ext_vector_type(4)))  float  v4f;   // clang vector: OK for
                                                           // __builtin_nontemporal_load

#define DH 128          // feature width
#define KTOT 256        // fused K (aggr 0..127, h 128..255)

__device__ __forceinline__ unsigned short f2bf(float f) {
    // round-to-nearest-even f32 -> bf16
    unsigned int u = __float_as_uint(f);
    unsigned int r = u + 0x7FFFu + ((u >> 16) & 1u);
    return (unsigned short)(r >> 16);
}

// ---------------------------------------------------------------------------
// Pack one layer's weights into the per-lane B-operand bf16 layout.
// B[k][o] = (k < 128) ? w_rel[o][k] : w_root[o][k-128]       (B is 256 x 128)
// Tile (kt 0..7, ot 0..7) is a 32x16 bf16 B operand. Per-lane pack:
//   lane l (l<16): n = l,  kk = j        (j = 0..15)
//   lane l (>=16): n = l-16, kk = 16 + j
// Blob: wpack[(kt*8+ot)*512 + lane*16 + j]   -> one contiguous 32B per lane.
// ---------------------------------------------------------------------------
__global__ __launch_bounds__(256)
void gcn_pack_weights(const float* __restrict__ w_rel,
                      const float* __restrict__ w_root,
                      unsigned short* __restrict__ wpack) {
    const int tile = blockIdx.x;          // 0..63
    const int kt = tile >> 3, ot = tile & 7;
    for (int e = threadIdx.x; e < 512; e += 256) {
        const int lane = e >> 4;
        const int j    = e & 15;
        const int kk   = (lane < 16) ? j : (16 + j);
        const int n    = lane & 15;
        const int K    = kt * 32 + kk;    // 0..255
        const int O    = ot * 16 + n;     // 0..127
        const float v  = (K < DH) ? w_rel[O * DH + K]
                                  : w_root[O * DH + (K - DH)];
        wpack[tile * 512 + e] = f2bf(v);
    }
}

// ---------------------------------------------------------------------------
// Edge scatter: one wave32 per edge. Edge id is forced wave-uniform via
// readfirstlane so src/dst come from scalar loads (KMcnt path). Lanes read a
// coalesced float4 slice of h[src] (512B/edge) and atomic-add into aggr[dst]
// (f32 atomics resolve in the 192MB L2 since aggr is 51MB).
// ---------------------------------------------------------------------------
__global__ __launch_bounds__(256)
void gcn_scatter(const float* __restrict__ h,
                 const long long* __restrict__ edge_index,
                 float* __restrict__ aggr,
                 int nEdges) {
    const int gw   = __builtin_amdgcn_readfirstlane(
                        (int)((blockIdx.x * 256u + threadIdx.x) >> 5));
    const int lane = threadIdx.x & 31;
    if (gw >= nEdges) return;
    const int src = (int)edge_index[gw];
    const int dst = (int)edge_index[(size_t)nEdges + gw];
    const v4f v = *(const v4f*)(h + (size_t)src * DH + lane * 4);
    float* b = aggr + (size_t)dst * DH + lane * 4;
    atomicAdd(b + 0, v[0]);
    atomicAdd(b + 1, v[1]);
    atomicAdd(b + 2, v[2]);
    atomicAdd(b + 3, v[3]);
}

// ---------------------------------------------------------------------------
// Fused dual-GEMM + bias + ReLU.
// Block = 256 threads = 8 wave32, handles 128 rows. Dynamic LDS (128KB):
//   smA (64KB): A = [aggr|h] rows as bf16 in a per-lane A-operand (same k order as smB)
//               layout: lane<16: M=lane, j0..15 -> K0..15
//                       lane>=16: M=lane-16, j0..15 -> K16..31
//   smB (64KB): the whole packed weight blob, staged once per block.
// Wave w owns M-tile w and sweeps 8 col-tiles x 8 K-tiles of
// v_wmma_f32_16x16x32_bf16 (64 WMMAs/wave), all operands from LDS.
// ---------------------------------------------------------------------------
__global__ __launch_bounds__(256)
void gcn_gemm(const float* __restrict__ aggr,
              const float* __restrict__ hin,
              const unsigned short* __restrict__ wpack,
              const float* __restrict__ bias,
              float* __restrict__ hout,
              int nNodes) {
    extern __shared__ unsigned short smem[];
    unsigned short* smA = smem;            // 8*8*32*16 ushorts = 64KB
    unsigned short* smB = smem + 32768;    // 64KB

    const int tid  = threadIdx.x;
    const int base = blockIdx.x * 128;
    const bool full = (base + 128) <= nNodes;

    // ---- stage B: copy packed weights (64KB) into LDS, 16B per op ----
    for (int idx = tid; idx < 4096; idx += 256) {
        ((uint4*)smB)[idx] = ((const uint4*)wpack)[idx];
    }

    // ---- stage A into LDS (coalesced, non-temporal global float4 reads) ----
    for (int idx = tid; idx < 8192; idx += 256) {      // 8192 float4 = 128x256 f32
        const int r   = idx >> 6;                      // row in block 0..127
        const int c4  = idx & 63;                      // float4 col 0..63
        const int row = base + r;
        v4f v = {0.f, 0.f, 0.f, 0.f};
        if (full || row < nNodes) {
            const float* s = (c4 < 32)
                ? (aggr + (size_t)row * DH + c4 * 4)
                : (hin  + (size_t)row * DH + (c4 - 32) * 4);
            v = __builtin_nontemporal_load((const v4f*)s);
        }
        const int mt = r >> 4, m = r & 15;
        const int k0 = c4 * 4;                         // 0..255, 4-aligned
        const int kt = k0 >> 5;
        const int kk = k0 & 31;
        const int lane = m + ((kk >> 4) << 4);
        const int j    = kk & 15;                          // 4 consecutive slots
        unsigned short* d = &smA[(((mt << 3) + kt) * 32 + lane) * 16 + j];
        d[0] = f2bf(v[0]); d[1] = f2bf(v[1]); d[2] = f2bf(v[2]); d[3] = f2bf(v[3]);
    }
    __syncthreads();

    // ---- WMMA sweep (EXEC all-ones throughout; only uniform branches) ----
    const int wv   = tid >> 5;    // M-tile
    const int lane = tid & 31;
    const int mt   = wv;

    for (int ot = 0; ot < 8; ++ot) {
        v8f c = {0.f, 0.f, 0.f, 0.f, 0.f, 0.f, 0.f, 0.f};
        const float bcol = bias[ot * 16 + (lane & 15)];
#pragma unroll
        for (int kt = 0; kt < 8; ++kt) {
            const v16bf a = *(const v16bf*)(smA +
                (((mt << 3) + kt) * 32 + lane) * 16);
            const v16bf b = *(const v16bf*)(smB +
                (((kt << 3) + ot) * 32 + lane) * 16);
            c = __builtin_amdgcn_wmma_f32_16x16x32_bf16(
                    false, a, false, b, (short)0, c, false, false);
        }
        // C layout: VGPR i, lanes 0-15 -> M=i, lanes 16-31 -> M=i+8; N=lane%16
        const int col   = ot * 16 + (lane & 15);
        const int rbase = base + mt * 16 + ((lane >> 4) << 3);
        if (full) {
#pragma unroll
            for (int i = 0; i < 8; ++i) {
                const float t = c[i] + bcol;
                hout[(size_t)(rbase + i) * DH + col] = t > 0.f ? t : 0.f;
            }
        } else {
#pragma unroll
            for (int i = 0; i < 8; ++i) {
                const int row = rbase + i;
                if (row < nNodes) {
                    const float t = c[i] + bcol;
                    hout[(size_t)row * DH + col] = t > 0.f ? t : 0.f;
                }
            }
        }
    }
}

// ---------------------------------------------------------------------------
// Head: out[n] = h[n] . w_out + b_out. One wave per node, shfl reduction.
// ---------------------------------------------------------------------------
__global__ __launch_bounds__(256)
void gcn_head(const float* __restrict__ h,
              const float* __restrict__ w_out,
              const float* __restrict__ b_out,
              float* __restrict__ out,
              int nNodes) {
    const int lane   = threadIdx.x & 31;
    const int gw     = (int)((blockIdx.x * 256u + threadIdx.x) >> 5);
    const int stride = (int)((gridDim.x * 256u) >> 5);
    const v4f  wv    = *(const v4f*)(w_out + lane * 4);
    const float bb   = b_out[0];
    for (int node = gw; node < nNodes; node += stride) {
        const v4f hv = __builtin_nontemporal_load(
                           (const v4f*)(h + (size_t)node * DH + lane * 4));
        float s = hv[0] * wv[0] + hv[1] * wv[1] + hv[2] * wv[2] + hv[3] * wv[3];
        s += __shfl_xor(s, 16);
        s += __shfl_xor(s, 8);
        s += __shfl_xor(s, 4);
        s += __shfl_xor(s, 2);
        s += __shfl_xor(s, 1);
        if (lane == 0) out[node] = s + bb;
    }
}

// ---------------------------------------------------------------------------
extern "C" void kernel_launch(void* const* d_in, const int* in_sizes, int n_in,
                              void* d_out, int out_size, void* d_ws, size_t ws_size,
                              hipStream_t stream) {
    const float*     x  = (const float*)d_in[0];
    const long long* ei = (const long long*)d_in[1];
    const float* w_rel[3]  = { (const float*)d_in[2], (const float*)d_in[5], (const float*)d_in[8]  };
    const float* b_rel[3]  = { (const float*)d_in[3], (const float*)d_in[6], (const float*)d_in[9]  };
    const float* w_root[3] = { (const float*)d_in[4], (const float*)d_in[7], (const float*)d_in[10] };
    const float* w_out = (const float*)d_in[11];
    const float* b_out = (const float*)d_in[12];
    float* out = (float*)d_out;

    const int nNodes = in_sizes[0] / DH;
    const int nEdges = in_sizes[1] / 2;

    // workspace: aggr | hA | hB | packed bf16 weights (3 x 64KB)
    const size_t nfeat = (size_t)nNodes * DH;
    float* aggr = (float*)d_ws;
    float* hA   = aggr + nfeat;
    float* hB   = hA + nfeat;
    unsigned short* wpack = (unsigned short*)(hB + nfeat);
    const size_t wlayer = 64 * 512;   // ushorts per layer

    // pack weights (runs every launch; deterministic)
    for (int l = 0; l < 3; ++l)
        gcn_pack_weights<<<64, 256, 0, stream>>>(w_rel[l], w_root[l],
                                                 wpack + l * wlayer);

    const int scatterBlocks = (nEdges + 7) / 8;        // 8 waves / block
    const int gemmBlocks    = (nNodes + 127) / 128;
    const size_t ldsBytes   = 2 * 32768 * sizeof(unsigned short);  // 128KB

    const float* hcur = x;
    float* houts[3] = { hA, hB, hA };
    for (int l = 0; l < 3; ++l) {
        (void)hipMemsetAsync(aggr, 0, nfeat * sizeof(float), stream);
        gcn_scatter<<<scatterBlocks, 256, 0, stream>>>(hcur, ei, aggr, nEdges);
        gcn_gemm<<<gemmBlocks, 256, ldsBytes, stream>>>(aggr, hcur,
                                                        wpack + l * wlayer,
                                                        b_rel[l], houts[l], nNodes);
        hcur = houts[l];
    }

    const int headBlocks = (nNodes + 7) / 8;           // 1 wave per node
    gcn_head<<<headBlocks, 256, 0, stream>>>(hcur, w_out, b_out, out, nNodes);
}
